// DigitConvolutionalModel_9758165696851
// MI455X (gfx1250) — matrix operand order repaired and checked by reading, not compiled, not run on hardware
//
#include <hip/hip_runtime.h>

typedef __attribute__((ext_vector_type(16))) _Float16 v16h;
typedef __attribute__((ext_vector_type(8)))  _Float16 v8h;
typedef __attribute__((ext_vector_type(8)))  float    v8f;

#define BT 32            // batch rows per workgroup
#define K1P 704          // 676 padded to 22*32
#define K1_TILES 22
#define K2_TILES 8       // K=256
#define NT1 16           // 256/16 n-tiles
#define FRAG_ELEMS 512   // 32 lanes * 16 f16 per B fragment

// ---------------------------------------------------------------------------
// Pack a row-major f32 weight matrix W[Kreal x Nstride] into f16 WMMA
// B-fragments. Fragment f = nt*Ktiles + kt holds 512 f16:
//   element(lane, e) = W[kt*32 + (e&7) + (e>=8 ? 16 : 0) + (lane<16 ? 0 : 8)][nt*16 + (lane&15)]
// zero-padded outside [Kreal, Nreal).
// ---------------------------------------------------------------------------
__global__ void pack_b_kernel(const float* __restrict__ W, _Float16* __restrict__ dst,
                              int Kreal, int Ktiles, int Nreal, int Nstride, int Ntiles) {
  int id = blockIdx.x * blockDim.x + threadIdx.x;
  int total = Ktiles * Ntiles * FRAG_ELEMS;
  if (id >= total) return;
  int f    = id >> 9;
  int r    = id & 511;
  int lane = r >> 4;
  int e    = r & 15;
  int nt   = f / Ktiles;
  int kt   = f - nt * Ktiles;
  int k = kt * 32 + (e & 7) + ((e >> 3) << 4) + ((lane >> 4) << 3);
  int n = nt * 16 + (lane & 15);
  float v = 0.f;
  if (k < Kreal && n < Nreal) v = W[(size_t)k * Nstride + n];
  dst[id] = (_Float16)v;
}

// Build a 16x32 f16 A-fragment from an LDS row pointer.
// lanes 0-15: slots 0..7 = K kk..kk+7,  slots 8..15 = K kk+16..kk+23
// lanes16-31: slots 0..7 = K kk+8..+15, slots 8..15 = K kk+24..+31
__device__ __forceinline__ v16h load_a_frag(const _Float16* row, int kk, int khalf) {
  v8h lo = *(const v8h*)(row + kk + khalf);
  v8h hi = *(const v8h*)(row + kk + 16 + khalf);
  return __builtin_shufflevector(lo, hi, 0,1,2,3,4,5,6,7,8,9,10,11,12,13,14,15);
}

__device__ __forceinline__ v16h load_b_frag(const _Float16* frag, int lane) {
  const v8h* p = (const v8h*)(frag + lane * 16);
  v8h lo = p[0];
  v8h hi = p[1];
  return __builtin_shufflevector(lo, hi, 0,1,2,3,4,5,6,7,8,9,10,11,12,13,14,15);
}

// ---------------------------------------------------------------------------
// Fused conv3x3(valid) -> [relu GEMM1 676->256] -> [relu GEMM2 256->256]
// -> [GEMM3 256->10]. One workgroup = BT=32 batch rows, 8 waves.
// ---------------------------------------------------------------------------
__global__ __launch_bounds__(256)
void fused_digit_mlp_kernel(const float* __restrict__ x,
                            const float* __restrict__ cw,
                            const _Float16* __restrict__ W1f,
                            const float* __restrict__ b1,
                            const _Float16* __restrict__ W2f,
                            const float* __restrict__ b2,
                            const _Float16* __restrict__ W3f,
                            const float* __restrict__ b3,
                            float* __restrict__ out) {
  // h0: BT x 704 f16 (45056 B) ; h1: BT x 256 f16 (16384 B)
  // h2 reuses h0 space (dead after GEMM1). Total static LDS = 61440 B.
  __shared__ __align__(16) _Float16 smem[BT * K1P + BT * 256];
  _Float16 (*h0)[K1P] = (_Float16(*)[K1P])smem;
  _Float16 (*h1)[256] = (_Float16(*)[256])(smem + BT * K1P);
  _Float16 (*h2)[256] = (_Float16(*)[256])smem;   // aliases h0

  const int tid  = threadIdx.x;
  const int lane = tid & 31;
  const int wave = tid >> 5;
  const int rowBase = blockIdx.x * BT;

  // ---- Stage A: 3x3 valid conv, f32 -> f16 into LDS (A-matrix operand) ----
  const float c00 = cw[0], c01 = cw[1], c02 = cw[2];
  const float c10 = cw[3], c11 = cw[4], c12 = cw[5];
  const float c20 = cw[6], c21 = cw[7], c22 = cw[8];

  // zero K-padding 676..703
  for (int i = tid; i < BT * (K1P - 676); i += 256) {
    int r = i / (K1P - 676);
    int p = i - r * (K1P - 676);
    h0[r][676 + p] = (_Float16)0.f;
  }
  for (int i = tid; i < BT * 676; i += 256) {
    int r  = i / 676;
    int p  = i - r * 676;
    int oy = p / 26;
    int ox = p - oy * 26;
    const float* xr = x + (size_t)(rowBase + r) * 784 + oy * 28 + ox;
    float s = xr[0]  * c00 + xr[1]  * c01 + xr[2]  * c02
            + xr[28] * c10 + xr[29] * c11 + xr[30] * c12
            + xr[56] * c20 + xr[57] * c21 + xr[58] * c22;
    h0[r][p] = (_Float16)s;
  }
  __syncthreads();

  const int mrow  = lane & 15;               // A row within 16-row tile
  const int khalf = (lane < 16) ? 0 : 8;     // A K-half select
  const int mofs  = (lane < 16) ? 0 : 8;     // D row offset
  const int ncol  = lane & 15;               // D column within 16-col tile

  // ---- Stage B: GEMM1  h1 = relu(h0[32x704] @ W1[704x256] + b1) ----
  for (int nn = 0; nn < 2; ++nn) {
    const int nt = wave + nn * 8;
    const _Float16* wf = W1f + (size_t)nt * K1_TILES * FRAG_ELEMS;
    v8f acc0 = {};
    v8f acc1 = {};
    for (int kt = 0; kt < K1_TILES; ++kt) {
      v16h b  = load_b_frag(wf + kt * FRAG_ELEMS, lane);
      v16h a0 = load_a_frag(&h0[mrow][0],      kt * 32, khalf);
      v16h a1 = load_a_frag(&h0[mrow + 16][0], kt * 32, khalf);
      acc0 = __builtin_amdgcn_wmma_f32_16x16x32_f16(false, a0, false, b, (short)0, acc0, false, false);
      acc1 = __builtin_amdgcn_wmma_f32_16x16x32_f16(false, a1, false, b, (short)0, acc1, false, false);
    }
    const int n = nt * 16 + ncol;
    const float bias = b1[n];
#pragma unroll
    for (int v = 0; v < 8; ++v) {
      float f0 = acc0[v] + bias; f0 = f0 > 0.f ? f0 : 0.f;
      float f1 = acc1[v] + bias; f1 = f1 > 0.f ? f1 : 0.f;
      h1[v + mofs][n]      = (_Float16)f0;
      h1[16 + v + mofs][n] = (_Float16)f1;
    }
  }
  __syncthreads();   // h0 now dead; h2 may overwrite it

  // ---- Stage C: GEMM2  h2 = relu(h1[32x256] @ W2[256x256] + b2) ----
  for (int nn = 0; nn < 2; ++nn) {
    const int nt = wave + nn * 8;
    const _Float16* wf = W2f + (size_t)nt * K2_TILES * FRAG_ELEMS;
    v8f acc0 = {};
    v8f acc1 = {};
#pragma unroll
    for (int kt = 0; kt < K2_TILES; ++kt) {
      v16h b  = load_b_frag(wf + kt * FRAG_ELEMS, lane);
      v16h a0 = load_a_frag(&h1[mrow][0],      kt * 32, khalf);
      v16h a1 = load_a_frag(&h1[mrow + 16][0], kt * 32, khalf);
      acc0 = __builtin_amdgcn_wmma_f32_16x16x32_f16(false, a0, false, b, (short)0, acc0, false, false);
      acc1 = __builtin_amdgcn_wmma_f32_16x16x32_f16(false, a1, false, b, (short)0, acc1, false, false);
    }
    const int n = nt * 16 + ncol;
    const float bias = b2[n];
#pragma unroll
    for (int v = 0; v < 8; ++v) {
      float f0 = acc0[v] + bias; f0 = f0 > 0.f ? f0 : 0.f;
      float f1 = acc1[v] + bias; f1 = f1 > 0.f ? f1 : 0.f;
      h2[v + mofs][n]      = (_Float16)f0;
      h2[16 + v + mofs][n] = (_Float16)f1;
    }
  }
  __syncthreads();

  // ---- Stage D: GEMM3  out = h2[32x256] @ W3[256x10] + b3 ----
  if (wave < 2) {
    const int mt = wave;
    v8f acc = {};
#pragma unroll
    for (int kt = 0; kt < K2_TILES; ++kt) {
      v16h b = load_b_frag(W3f + kt * FRAG_ELEMS, lane);
      v16h a = load_a_frag(&h2[mt * 16 + mrow][0], kt * 32, khalf);
      acc = __builtin_amdgcn_wmma_f32_16x16x32_f16(false, a, false, b, (short)0, acc, false, false);
    }
    if (ncol < 10) {
      const float bias = b3[ncol];
#pragma unroll
      for (int v = 0; v < 8; ++v) {
        out[(size_t)(rowBase + mt * 16 + mofs + v) * 10 + ncol] = acc[v] + bias;
      }
    }
  }
}

extern "C" void kernel_launch(void* const* d_in, const int* in_sizes, int n_in,
                              void* d_out, int out_size, void* d_ws, size_t ws_size,
                              hipStream_t stream) {
  const float* x  = (const float*)d_in[0];
  const float* cw = (const float*)d_in[1];
  const float* W1 = (const float*)d_in[2];
  const float* b1 = (const float*)d_in[3];
  const float* W2 = (const float*)d_in[4];
  const float* b2 = (const float*)d_in[5];
  const float* W3 = (const float*)d_in[6];
  const float* b3 = (const float*)d_in[7];
  float* out = (float*)d_out;

  const int B = in_sizes[0] / 784;   // 16384

  // workspace: packed f16 B-fragments for W1 / W2 / W3
  _Float16* W1f = (_Float16*)d_ws;                                   // 22*16*512 = 180224 f16
  _Float16* W2f = W1f + (size_t)K1_TILES * NT1 * FRAG_ELEMS;         //  8*16*512 =  65536 f16
  _Float16* W3f = W2f + (size_t)K2_TILES * NT1 * FRAG_ELEMS;         //  8* 1*512 =   4096 f16
  // total ~500 KB of workspace

  {
    int tot = K1_TILES * NT1 * FRAG_ELEMS;
    pack_b_kernel<<<(tot + 255) / 256, 256, 0, stream>>>(W1, W1f, 676, K1_TILES, 256, 256, NT1);
  }
  {
    int tot = K2_TILES * NT1 * FRAG_ELEMS;
    pack_b_kernel<<<(tot + 255) / 256, 256, 0, stream>>>(W2, W2f, 256, K2_TILES, 256, 256, NT1);
  }
  {
    int tot = K2_TILES * 1 * FRAG_ELEMS;
    pack_b_kernel<<<(tot + 255) / 256, 256, 0, stream>>>(W3, W3f, 256, K2_TILES, 10, 10, 1);
  }

  fused_digit_mlp_kernel<<<B / BT, 256, 0, stream>>>(x, cw, W1f, b1, W2f, b2, W3f, b3, out);
}
